// SelfAttention_layer_23751169147174
// MI455X (gfx1250) — matrix operand order repaired and checked by reading, not compiled, not run on hardware
//
#include <hip/hip_runtime.h>

#define BB 16
#define CC 128
#define DD 2048
static constexpr size_t SZ = (size_t)BB * CC * DD;        // elems per [B,C,D] tensor
static constexpr float ATT_SCALE = 0.088388347648318447f; // 1/sqrt(128)

typedef __bf16 bf16_t;
typedef __attribute__((ext_vector_type(16))) __bf16 v16bf;
typedef __attribute__((ext_vector_type(8)))  float  v8f;

__device__ __forceinline__ v8f wmma_bf16(v16bf a, v16bf b, v8f c) {
  // D = A(16x32 bf16) * B(32x16 bf16) + C(16x16 f32)
  return __builtin_amdgcn_wmma_f32_16x16x32_bf16(false, a, false, b, (short)0, c,
                                                 false, false);
}

// A-fragment (16x32 bf16): lane half h, vgpr-pair i -> K index of low element
__device__ __forceinline__ int kpatA(int i, int h) {
  return (i < 4) ? (8 * h + 2 * i) : (16 + 8 * h + 2 * (i - 4));
}
// B-fragment (32x16 bf16): K index of low element
__device__ __forceinline__ int kpatB(int i, int h) { return (i < 4) ? (8 * h + 2 * i) : (16 + 8 * h + 2 * (i - 4)); }

// ---------------------------------------------------------------------------
// Kernel 1: yZ = W_z @ x_b + bias_z for z in {f,g,h}; fp32 results to ws.
// Per block: M=128 (4 waves x 32 rows), N=64, K=128 in 4 chunks of 32.
// ---------------------------------------------------------------------------
__global__ __launch_bounds__(128) void conv3_kernel(
    const float* __restrict__ x, const float* __restrict__ Wf,
    const float* __restrict__ bfp, const float* __restrict__ Wg,
    const float* __restrict__ bgp, const float* __restrict__ Wh,
    const float* __restrict__ bhp, float* __restrict__ ws) {
  const int b = blockIdx.y;
  const int z = blockIdx.z;
  const int d0 = blockIdx.x * 64;
  const float* W = (z == 0) ? Wf : (z == 1) ? Wg : Wh;
  const float* bias = (z == 0) ? bfp : (z == 1) ? bgp : bhp;
  float* y = ws + (size_t)z * SZ;

  __shared__ bf16_t Wl[128][32]; // [m][k]  8KB
  __shared__ bf16_t Xl[64][32];  // [d][k]  4KB

  const int t = threadIdx.x;
  const int wave = t >> 5;
  const int lane = t & 31;
  const int ln = lane & 15;
  const int h = lane >> 4;
  const int m0 = wave * 32;

  v8f acc[2][4];
  for (int mi = 0; mi < 2; mi++)
    for (int nj = 0; nj < 4; nj++)
      for (int i = 0; i < 8; i++) acc[mi][nj][i] = 0.0f;

  for (int kc = 0; kc < 4; kc++) {
    const int c0 = kc * 32;
    { // stage W chunk: thread t = row m, 32 k's
      const float* wr = W + (size_t)t * CC + c0;
#pragma unroll
      for (int k = 0; k < 32; k++) Wl[t][k] = (bf16_t)wr[k];
    }
    { // stage x chunk: c_l = t>>2 (0..31), d_l = 16*(t&3)..+15
      const int c_l = t >> 2, q = t & 3;
      const float* xr = x + ((size_t)b * CC + (c0 + c_l)) * DD + d0 + q * 16;
#pragma unroll
      for (int k = 0; k < 16; k++) Xl[q * 16 + k][c_l] = (bf16_t)xr[k];
    }
    __syncthreads();

    v16bf afr[2];
#pragma unroll
    for (int mi = 0; mi < 2; mi++) {
      const int m = m0 + 16 * mi + ln;
#pragma unroll
      for (int i = 0; i < 8; i++) {
        const int K = kpatA(i, h);
        afr[mi][2 * i] = Wl[m][K];
        afr[mi][2 * i + 1] = Wl[m][K + 1];
      }
    }
#pragma unroll
    for (int nj = 0; nj < 4; nj++) {
      v16bf bfr;
      const int dl = 16 * nj + ln;
#pragma unroll
      for (int i = 0; i < 8; i++) {
        const int K = kpatB(i, h);
        bfr[2 * i] = Xl[dl][K];
        bfr[2 * i + 1] = Xl[dl][K + 1];
      }
      acc[0][nj] = wmma_bf16(afr[0], bfr, acc[0][nj]);
      acc[1][nj] = wmma_bf16(afr[1], bfr, acc[1][nj]);
    }
    __syncthreads();
  }

#pragma unroll
  for (int mi = 0; mi < 2; mi++)
#pragma unroll
    for (int nj = 0; nj < 4; nj++)
#pragma unroll
      for (int i = 0; i < 8; i++) {
        const int m = m0 + 16 * mi + i + 8 * h;
        const int d = d0 + 16 * nj + ln;
        y[((size_t)b * CC + m) * DD + d] = acc[mi][nj][i] + bias[m];
      }
}

// ---------------------------------------------------------------------------
// Kernel 2: per-channel BN stats over (B,D); fold with gamma/beta into
// scale/shift.  scsh layout: [scf 128][shf 128][scg 128][shg 128]
// ---------------------------------------------------------------------------
__global__ __launch_bounds__(256) void stats_kernel(
    const float* __restrict__ ws, const float* __restrict__ gamma_f,
    const float* __restrict__ beta_f, const float* __restrict__ gamma_g,
    const float* __restrict__ beta_g, float* __restrict__ scsh) {
  const int c = blockIdx.x;
  const int which = blockIdx.y; // 0=f, 1=g
  const float* y = ws + (size_t)which * SZ;
  const float* gamma = which ? gamma_g : gamma_f;
  const float* beta = which ? beta_g : beta_f;
  float s = 0.0f, q = 0.0f;
  for (int j = threadIdx.x; j < BB * DD; j += 256) {
    const int b = j >> 11, d = j & 2047;
    const float v = y[((size_t)b * CC + c) * DD + d];
    s += v;
    q += v * v;
  }
  __shared__ float sbuf[256], qbuf[256];
  sbuf[threadIdx.x] = s;
  qbuf[threadIdx.x] = q;
  __syncthreads();
  for (int o = 128; o > 0; o >>= 1) {
    if ((int)threadIdx.x < o) {
      sbuf[threadIdx.x] += sbuf[threadIdx.x + o];
      qbuf[threadIdx.x] += qbuf[threadIdx.x + o];
    }
    __syncthreads();
  }
  if (threadIdx.x == 0) {
    const float n = (float)(BB * DD);
    const float mean = sbuf[0] / n;
    const float var = qbuf[0] / n - mean * mean; // biased (torch default)
    const float inv = rsqrtf(var + 1e-5f);
    const float sc = gamma[c] * inv;
    scsh[which * 256 + c] = sc;
    scsh[which * 256 + 128 + c] = beta[c] - mean * sc;
  }
}

// ---------------------------------------------------------------------------
// Kernel 3: fused flash attention + output conv + residual, software-pipelined.
// Block = 8 waves; wave w owns d rows [blk*128 + 16w, +16).  Streams e in
// 32-col steps with online softmax and double-buffered LDS staging: global
// loads for step t+1 issue at the top of step t and are hidden behind 16
// WMMAs + softmax VALU work.  attn matrix (268MB > L2) never materialized.
// ---------------------------------------------------------------------------
__global__ __launch_bounds__(256) void attn_kernel(
    const float* __restrict__ ws, const float* __restrict__ scsh,
    const float* __restrict__ Wv, const float* __restrict__ bv,
    const float* __restrict__ x, float* __restrict__ out) {
  const int b = blockIdx.y;
  const int t = threadIdx.x;
  const int wave = t >> 5;
  const int lane = t & 31;
  const int ln = lane & 15;
  const int h = lane >> 4;
  const int d0 = blockIdx.x * 128 + wave * 16;

  const float* yf = ws;
  const float* yg = ws + SZ;
  const float* yh = ws + 2 * SZ;

  // Phase-1 view: gs[2][32][128] (16KB) | hs[2][128][32] (16KB) | ps[8][16][32] (8KB)
  // Phase-2 view: os[8][16][128] (32KB) overlays gs/hs (disjoint in time).
  __shared__ __align__(16) char smem[40 * 1024];
  bf16_t(*gs)[32][128] = (bf16_t(*)[32][128])(smem);           // [e][c]
  bf16_t(*hs)[128][32] = (bf16_t(*)[128][32])(smem + 16384);   // [c][e]
  bf16_t(*ps)[16][32] = (bf16_t(*)[16][32])(smem + 32768);     // per-wave P
  bf16_t(*os)[16][128] = (bf16_t(*)[16][128])(smem);           // per-wave O

  // Staging geometry: thread -> (channel c2, 16-element half-run of e)
  const int c2 = t >> 1, half = t & 1;
  const float gsc = scsh[256 + c2], gsh = scsh[384 + c2];
  const float* grow = yg + ((size_t)b * CC + c2) * DD + half * 16;
  const float* hrow = yh + ((size_t)b * CC + c2) * DD + half * 16;

  // Preload A fragments of f^T (rows d, K=c), BN+ReLU+softmax-scale folded in.
  v16bf af[4];
#pragma unroll
  for (int kc = 0; kc < 4; kc++) {
#pragma unroll
    for (int i = 0; i < 8; i++) {
      const int c = 32 * kc + kpatA(i, h);
      float v0 = yf[((size_t)b * CC + c) * DD + (d0 + ln)];
      float v1 = yf[((size_t)b * CC + c + 1) * DD + (d0 + ln)];
      v0 = fmaxf(scsh[c] * v0 + scsh[128 + c], 0.0f) * ATT_SCALE;
      v1 = fmaxf(scsh[c + 1] * v1 + scsh[128 + c + 1], 0.0f) * ATT_SCALE;
      af[kc][2 * i] = (bf16_t)v0;
      af[kc][2 * i + 1] = (bf16_t)v1;
    }
  }

  float m[8], sacc[8];
  v8f O[8];
#pragma unroll
  for (int i = 0; i < 8; i++) {
    m[i] = -3.0e38f;
    sacc[i] = 0.0f;
    for (int j = 0; j < 8; j++) O[j][i] = 0.0f;
  }

  constexpr int NE = DD / 32;

  { // prologue: stage chunk 0 into buffer 0
    const float4* g4 = (const float4*)(grow);
    const float4* h4 = (const float4*)(hrow);
#pragma unroll
    for (int v = 0; v < 4; v++) {
      const float4 gv = g4[v], hv = h4[v];
      const float gg[4] = {gv.x, gv.y, gv.z, gv.w};
      const float hh[4] = {hv.x, hv.y, hv.z, hv.w};
#pragma unroll
      for (int k = 0; k < 4; k++) {
        gs[0][half * 16 + 4 * v + k][c2] =
            (bf16_t)fmaxf(gsc * gg[k] + gsh, 0.0f);
        hs[0][c2][half * 16 + 4 * v + k] = (bf16_t)hh[k];
      }
    }
  }
  __syncthreads();

#pragma unroll 1
  for (int et = 0; et < NE; et++) {
    const int cur = et & 1, nxt = cur ^ 1;
    const bool have_next = (et + 1) < NE;

    // (1) issue next chunk's global loads; latency hidden by this iteration
    float4 gv[4], hv[4];
    if (have_next) {
      const int en = (et + 1) * 32;
      const float4* g4 = (const float4*)(grow + en);
      const float4* h4 = (const float4*)(hrow + en);
#pragma unroll
      for (int v = 0; v < 4; v++) { gv[v] = g4[v]; hv[v] = h4[v]; }
    }

    // (2) S = f^T g  (two 16x16 f32 fragments: e cols 0..15 and 16..31)
    v8f S0, S1;
#pragma unroll
    for (int i = 0; i < 8; i++) { S0[i] = 0.0f; S1[i] = 0.0f; }
#pragma unroll
    for (int kc = 0; kc < 4; kc++) {
      v16bf b0, b1;
#pragma unroll
      for (int i = 0; i < 8; i++) {
        const int c = 32 * kc + kpatB(i, h);
        b0[2 * i] = gs[cur][ln][c];
        b0[2 * i + 1] = gs[cur][ln][c + 1];
        b1[2 * i] = gs[cur][ln + 16][c];
        b1[2 * i + 1] = gs[cur][ln + 16][c + 1];
      }
      S0 = wmma_bf16(af[kc], b0, S0);
      S1 = wmma_bf16(af[kc], b1, S1);
    }

    // (3) online softmax.  C-layout: comp i is row (i+8h); cols across lanes.
    float alpha[8];
#pragma unroll
    for (int i = 0; i < 8; i++) {
      float mx = fmaxf(S0[i], S1[i]);
#pragma unroll
      for (int off = 1; off < 16; off <<= 1)
        mx = fmaxf(mx, __shfl_xor(mx, off, 16));
      const float nm = fmaxf(m[i], mx);
      alpha[i] = __expf(m[i] - nm);
      m[i] = nm;
      const float p0 = __expf(S0[i] - nm);
      const float p1 = __expf(S1[i] - nm);
      S0[i] = p0;
      S1[i] = p1;
      float rs = p0 + p1;
#pragma unroll
      for (int off = 1; off < 16; off <<= 1) rs += __shfl_xor(rs, off, 16);
      sacc[i] = sacc[i] * alpha[i] + rs;
    }
#pragma unroll
    for (int j = 0; j < 8; j++)
#pragma unroll
      for (int i = 0; i < 8; i++) O[j][i] *= alpha[i];

    // (4) relayout P: C-layout f32 -> A-layout bf16 via per-wave LDS tile
#pragma unroll
    for (int i = 0; i < 8; i++) {
      ps[wave][i + 8 * h][ln] = (bf16_t)S0[i];
      ps[wave][i + 8 * h][ln + 16] = (bf16_t)S1[i];
    }
    __syncthreads(); // (A) ps visible
    v16bf pa;
#pragma unroll
    for (int i = 0; i < 8; i++) {
      const int K = kpatA(i, h);
      pa[2 * i] = ps[wave][ln][K];
      pa[2 * i + 1] = ps[wave][ln][K + 1];
    }

    // (5) O += P @ hT   (16d x 32e) * (32e x 128c)
#pragma unroll
    for (int j = 0; j < 8; j++) {
      v16bf hb;
      const int c = 16 * j + ln;
#pragma unroll
      for (int i = 0; i < 8; i++) {
        const int K = kpatB(i, h);
        hb[2 * i] = hs[cur][c][K];
        hb[2 * i + 1] = hs[cur][c][K + 1];
      }
      O[j] = wmma_bf16(pa, hb, O[j]);
    }

    // (6) convert + store next chunk into the other buffer
    if (have_next) {
#pragma unroll
      for (int v = 0; v < 4; v++) {
        const float gg[4] = {gv[v].x, gv[v].y, gv[v].z, gv[v].w};
        const float hh[4] = {hv[v].x, hv[v].y, hv[v].z, hv[v].w};
#pragma unroll
        for (int k = 0; k < 4; k++) {
          gs[nxt][half * 16 + 4 * v + k][c2] =
              (bf16_t)fmaxf(gsc * gg[k] + gsh, 0.0f);
          hs[nxt][c2][half * 16 + 4 * v + k] = (bf16_t)hh[k];
        }
      }
    }
    __syncthreads(); // (B) next buffer ready; cur reads complete
  }

  // Normalize rows, then fuse output conv: outT = O @ Wv^T (+ bv + x).
#pragma unroll
  for (int i = 0; i < 8; i++) {
    const float inv = 1.0f / sacc[i];
#pragma unroll
    for (int j = 0; j < 8; j++) O[j][i] *= inv;
  }
#pragma unroll
  for (int j = 0; j < 8; j++)
#pragma unroll
    for (int i = 0; i < 8; i++)
      os[wave][i + 8 * h][16 * j + ln] = (bf16_t)O[j][i];
  __syncthreads();

  v8f R[8];
#pragma unroll
  for (int j = 0; j < 8; j++)
#pragma unroll
    for (int i = 0; i < 8; i++) R[j][i] = 0.0f;
#pragma unroll
  for (int kc = 0; kc < 4; kc++) {
    v16bf aa;
#pragma unroll
    for (int i = 0; i < 8; i++) {
      const int K = 32 * kc + kpatA(i, h);
      aa[2 * i] = os[wave][ln][K];
      aa[2 * i + 1] = os[wave][ln][K + 1];
    }
#pragma unroll
    for (int j = 0; j < 8; j++) {
      v16bf wb;
      const int o = 16 * j + ln;
#pragma unroll
      for (int i = 0; i < 8; i++) {
        const int c = 32 * kc + kpatB(i, h);
        wb[2 * i] = (bf16_t)Wv[(size_t)o * CC + c]; // Wv^T[c][o]
        wb[2 * i + 1] = (bf16_t)Wv[(size_t)o * CC + c + 1];
      }
      R[j] = wmma_bf16(aa, wb, R[j]);
    }
  }
#pragma unroll
  for (int j = 0; j < 8; j++) {
    const int o = 16 * j + ln;
    const float bvo = bv[o];
#pragma unroll
    for (int i = 0; i < 8; i++) {
      const int d = d0 + i + 8 * h;
      const size_t idx = ((size_t)b * CC + o) * DD + d;
      out[idx] = R[j][i] + bvo + x[idx];
    }
  }
}

// ---------------------------------------------------------------------------
extern "C" void kernel_launch(void* const* d_in, const int* in_sizes, int n_in,
                              void* d_out, int out_size, void* d_ws,
                              size_t ws_size, hipStream_t stream) {
  (void)in_sizes; (void)n_in; (void)out_size; (void)ws_size;
  const float* x = (const float*)d_in[0];
  const float* Wf = (const float*)d_in[1];
  const float* bfp = (const float*)d_in[2];
  const float* gamma_f = (const float*)d_in[3];
  const float* beta_f = (const float*)d_in[4];
  const float* Wg = (const float*)d_in[5];
  const float* bgp = (const float*)d_in[6];
  const float* gamma_g = (const float*)d_in[7];
  const float* beta_g = (const float*)d_in[8];
  const float* Wh = (const float*)d_in[9];
  const float* bhp = (const float*)d_in[10];
  const float* Wv = (const float*)d_in[11];
  const float* bv = (const float*)d_in[12];
  float* ws = (float*)d_ws;
  float* out = (float*)d_out;
  float* scsh = ws + 3 * SZ; // 512 floats of folded BN scale/shift

  conv3_kernel<<<dim3(DD / 64, BB, 3), 128, 0, stream>>>(x, Wf, bfp, Wg, bgp,
                                                         Wh, bhp, ws);
  stats_kernel<<<dim3(CC, 2), 256, 0, stream>>>(ws, gamma_f, beta_f, gamma_g,
                                                beta_g, scsh);
  attn_kernel<<<dim3(DD / 128, BB), 256, 0, stream>>>(ws, scsh, Wv, bv, x, out);
}
